// CrossModalVSSBlock_57277683859940
// MI455X (gfx1250) — matrix operand order repaired and checked by reading, compiled, not run on hardware
//
#include <hip/hip_runtime.h>
#include <math.h>

typedef __attribute__((ext_vector_type(16))) _Float16 v16h;
typedef __attribute__((ext_vector_type(8)))  _Float16 v8h;
typedef __attribute__((ext_vector_type(8)))  float    v8f;

// =====================================================================
// WMMA GEMM:  C[M,N] (f32) = A[M,K] (f16, row-major, lda)
//                           x W[N,K] (f16, row-major, ldb)^T   (+ bias)
// One wave32 per 16x16 output tile, K stepped by 32 via
// v_wmma_f32_16x16x32_f16.  M %16==0, N %16==0, K %32==0 (padded in ws).
// Fragment layouts per CDNA5 ISA 7.12.2 (16-bit A 16x32, B 32x16, f32 C/D).
// =====================================================================
__global__ void k_gemm_wmma(const _Float16* __restrict__ A,
                            const _Float16* __restrict__ Bw,
                            float* __restrict__ C,
                            const float* __restrict__ bias,
                            int N, int K, int lda, int ldb, int ldc,
                            int tilesN, int ntiles)
{
    const int tile = blockIdx.x * (blockDim.x >> 5) + (threadIdx.x >> 5);
    if (tile >= ntiles) return;               // whole-wave exit: EXEC stays all-1s
    const int tm = tile / tilesN, tn = tile % tilesN;
    const int lane = threadIdx.x & 31;
    const int hl = lane >> 4;                 // K half-group select
    const int ll = lane & 15;
    const _Float16* arow = A  + (size_t)(tm * 16 + ll) * lda + hl * 8;
    const _Float16* brow = Bw + (size_t)(tn * 16 + ll) * ldb + hl * 8;
    v8f acc = {0.f, 0.f, 0.f, 0.f, 0.f, 0.f, 0.f, 0.f};
    for (int k0 = 0; k0 < K; k0 += 32) {
        v16h a, b;
        // A lane ll (row M): K = k0+hl*8 .. +7  and  k0+hl*8+16 .. +23
        ((v8h*)&a)[0] = *(const v8h*)(arow + k0);
        ((v8h*)&a)[1] = *(const v8h*)(arow + k0 + 16);
        // B lane ll (col N): K = k0+hl*8 .. +7  and  k0+hl*8+16 .. +23
        ((v8h*)&b)[0] = *(const v8h*)(brow + k0);
        ((v8h*)&b)[1] = *(const v8h*)(brow + k0 + 16);
        if (k0 + 32 < K) {                    // near-scope prefetch of next K tile
            __builtin_prefetch(arow + k0 + 32, 0, 3);
            __builtin_prefetch(brow + k0 + 32, 0, 3);
        }
        acc = __builtin_amdgcn_wmma_f32_16x16x32_f16(
            /*neg_a=*/false, a, /*neg_b=*/false, b,
            /*c_mod=*/(short)0, acc, /*reuse_a=*/false, /*reuse_b=*/false);
    }
    const int n  = tn * 16 + ll;
    const float bv = bias ? bias[n] : 0.f;
    const int m0 = tm * 16 + hl * 8;          // C/D: VGPR r -> row m0+r
#pragma unroll
    for (int r = 0; r < 8; ++r)
        C[(size_t)(m0 + r) * ldc + n] = acc[r] + bv;
}

// ---------------- f32 -> f16 convert with zero padding -------------------
// dst[Np,Kp] <- src[Nr,Kr]; out-of-range filled with 0.
__global__ void k_w2h(const float* __restrict__ src, _Float16* __restrict__ dst,
                      int Nr, int Kr, int Np, int Kp)
{
    int i = blockIdx.x * blockDim.x + threadIdx.x;
    if (i >= Np * Kp) return;
    int n = i / Kp, k = i % Kp;
    dst[i] = (n < Nr && k < Kr) ? (_Float16)src[(size_t)n * Kr + k] : (_Float16)0.f;
}

// ---------------- LayerNorm over last dim, one wave per row --------------
__global__ void k_ln(const float* __restrict__ x, const float* __restrict__ g,
                     const float* __restrict__ b, float* __restrict__ out,
                     int M, int C, float eps)
{
    int row  = blockIdx.x * (blockDim.x >> 5) + (threadIdx.x >> 5);
    int lane = threadIdx.x & 31;
    if (row >= M) return;
    const float* xr = x + (size_t)row * C;
    float s = 0.f;
    for (int c = lane; c < C; c += 32) s += xr[c];
    for (int o = 16; o; o >>= 1) s += __shfl_xor(s, o, 32);
    float mean = s / (float)C;
    float v = 0.f;
    for (int c = lane; c < C; c += 32) { float t = xr[c] - mean; v += t * t; }
    for (int o = 16; o; o >>= 1) v += __shfl_xor(v, o, 32);
    float inv = rsqrtf(v / (float)C + eps);
    float* orow = out + (size_t)row * C;
    for (int c = lane; c < C; c += 32)
        orow[c] = (xr[c] - mean) * inv * g[c] + b[c];
}

// ---------------- depthwise 3x3 conv (SAME) + SiLU -----------------------
__global__ void k_dwconv(const float* __restrict__ xz, const float* __restrict__ wc,
                         const float* __restrict__ bc, float* __restrict__ xc,
                         int Bb, int H, int W, int d, int ld2)
{
    int i = blockIdx.x * blockDim.x + threadIdx.x;
    int L = H * W;
    if (i >= Bb * L * d) return;
    int di = i % d; int t = i / d; int l = t % L; int b = t / L;
    int h = l / W, wq = l % W;
    const float* wp = wc + (size_t)di * 9;
    float s = bc[di];
#pragma unroll
    for (int dh = -1; dh <= 1; ++dh) {
        int hh = h + dh; if ((unsigned)hh >= (unsigned)H) continue;
#pragma unroll
        for (int dw = -1; dw <= 1; ++dw) {
            int ww = wq + dw; if ((unsigned)ww >= (unsigned)W) continue;
            s += xz[((size_t)b * L + (size_t)hh * W + ww) * ld2 + di] *
                 wp[(dh + 1) * 3 + (dw + 1)];
        }
    }
    s = s / (1.f + __expf(-s));               // SiLU
    xc[i] = s;
}

// ---------------- build 4 scan-direction copies (f32 + f16) --------------
__global__ void k_build_xs(const float* __restrict__ xc, float* __restrict__ xs,
                           _Float16* __restrict__ xsh, int Bb, int H, int W, int d)
{
    int i = blockIdx.x * blockDim.x + threadIdx.x;
    int L = H * W;
    if (i >= Bb * L * d) return;
    int di = i % d; int t = i / d; int l = t % L; int b = t / L;
    int h = l / W, wq = l % W;
    float v = xc[i];
    size_t Md = (size_t)Bb * L * d;
    size_t base = (size_t)b * L * d;
    int lt = wq * H + h;
    size_t i0 = 0 * Md + base + (size_t)l * d + di;
    size_t i1 = 1 * Md + base + (size_t)lt * d + di;
    size_t i2 = 2 * Md + base + (size_t)(L - 1 - l) * d + di;
    size_t i3 = 3 * Md + base + (size_t)(L - 1 - lt) * d + di;
    xs[i0] = v; xs[i1] = v; xs[i2] = v; xs[i3] = v;
    _Float16 vh = (_Float16)v;
    xsh[i0] = vh; xsh[i1] = vh; xsh[i2] = vh; xsh[i3] = vh;
}

// =====================================================================
// Chunked (3-phase) associative scan.  h_l = a_l*h_{l-1} + b_l with
// a_l = exp(dt*A_j), b_l = dt*u*B_l[j].  L split into S chunks:
//   phase1: per (channel,chunk) compute composed (P,Q): h_out = P*h_in + Q
//   phase2: per channel, sequential compose over S chunks -> entry states
//   phase3: per (channel,chunk) replay from entry state, emit y (+D*u)
// Turns a 4096-long serial chain into 128 steps with 32x thread count.
// =====================================================================
template <int NS>
__global__ void k_scan_p1(const float* __restrict__ u, const float* __restrict__ xdbl,
                          const float* __restrict__ dts, const float* __restrict__ dtb,
                          const float* __restrict__ alog,
                          float* __restrict__ Pb, float* __restrict__ Qb,
                          int Bb, int L, int d, int ldx, int R, int S)
{
    int channels = Bb * 4 * d;
    int idx = blockIdx.x * blockDim.x + threadIdx.x;
    if (idx >= channels * S) return;
    int ch = idx % channels;                  // adjacent threads: adjacent d -> coalesced
    int c  = idx / channels;
    int di = ch % d; int kb = ch / d; int b = kb % Bb; int k = kb / Bb;
    int T = L / S, l0 = c * T;
    float P[NS], Q[NS], Aa[NS];
#pragma unroll
    for (int j = 0; j < NS; ++j) {
        P[j] = 1.f; Q[j] = 0.f;
        Aa[j] = -__expf(alog[((size_t)k * d + di) * NS + j]);
    }
    float bias = dtb[(size_t)k * d + di];
    size_t cb = (size_t)(k * Bb + b) * L;
    const float* ub = u    + cb * d + di;
    const float* db = dts  + cb * d + di;
    const float* xb = xdbl + cb * ldx;
    for (int l = l0; l < l0 + T; ++l) {
        float uu = ub[(size_t)l * d];
        float t  = db[(size_t)l * d] + bias;
        float dt = (t > 20.f) ? t : log1pf(__expf(t));   // softplus
        float du = dt * uu;
        const float* Bt = xb + (size_t)l * ldx + R;
#pragma unroll
        for (int j = 0; j < NS; ++j) {
            float a = __expf(dt * Aa[j]);
            P[j] *= a;
            Q[j] = a * Q[j] + du * Bt[j];
        }
    }
    float* pp = Pb + ((size_t)c * channels + ch) * NS;
    float* qq = Qb + ((size_t)c * channels + ch) * NS;
#pragma unroll
    for (int j = 0; j < NS; ++j) { pp[j] = P[j]; qq[j] = Q[j]; }
}

template <int NS>
__global__ void k_scan_p2(const float* __restrict__ Pb, const float* __restrict__ Qb,
                          float* __restrict__ Hin, int channels, int S)
{
    int ch = blockIdx.x * blockDim.x + threadIdx.x;
    if (ch >= channels) return;
    float h[NS];
#pragma unroll
    for (int j = 0; j < NS; ++j) h[j] = 0.f;
    for (int c = 0; c < S; ++c) {
        size_t o = ((size_t)c * channels + ch) * NS;
#pragma unroll
        for (int j = 0; j < NS; ++j) {
            Hin[o + j] = h[j];                 // entry state for chunk c
            h[j] = Pb[o + j] * h[j] + Qb[o + j];
        }
    }
}

template <int NS>
__global__ void k_scan_p3(const float* __restrict__ u, const float* __restrict__ xdbl,
                          const float* __restrict__ dts, const float* __restrict__ dtb,
                          const float* __restrict__ alog, const float* __restrict__ Dp,
                          const float* __restrict__ Hin, float* __restrict__ y,
                          int Bb, int L, int d, int ldx, int R, int S)
{
    int channels = Bb * 4 * d;
    int idx = blockIdx.x * blockDim.x + threadIdx.x;
    if (idx >= channels * S) return;
    int ch = idx % channels;
    int c  = idx / channels;
    int di = ch % d; int kb = ch / d; int b = kb % Bb; int k = kb / Bb;
    int T = L / S, l0 = c * T;
    float hst[NS], Aa[NS];
    const float* hh = Hin + ((size_t)c * channels + ch) * NS;
#pragma unroll
    for (int j = 0; j < NS; ++j) {
        hst[j] = hh[j];
        Aa[j] = -__expf(alog[((size_t)k * d + di) * NS + j]);
    }
    float bias = dtb[(size_t)k * d + di];
    float Dv   = Dp[(size_t)k * d + di];
    size_t cb = (size_t)(k * Bb + b) * L;
    const float* ub = u    + cb * d + di;
    const float* db = dts  + cb * d + di;
    const float* xb = xdbl + cb * ldx;
    float* yb = y + cb * d + di;
    for (int l = l0; l < l0 + T; ++l) {
        float uu = ub[(size_t)l * d];
        float t  = db[(size_t)l * d] + bias;
        float dt = (t > 20.f) ? t : log1pf(__expf(t));
        float du = dt * uu;
        const float* Bt = xb + (size_t)l * ldx + R;
        const float* Ct = Bt + NS;
        float yv = 0.f;
#pragma unroll
        for (int j = 0; j < NS; ++j) {
            hst[j] = __expf(dt * Aa[j]) * hst[j] + du * Bt[j];
            yv += hst[j] * Ct[j];
        }
        yb[(size_t)l * d] = yv + Dv * uu;                // + D*u folded in
    }
}

// ---------------- merge 4 scan directions back to row-major --------------
__global__ void k_merge(const float* __restrict__ y, float* __restrict__ ym,
                        int Bb, int H, int W, int d)
{
    int i = blockIdx.x * blockDim.x + threadIdx.x;
    int L = H * W;
    if (i >= Bb * L * d) return;
    int di = i % d; int t = i / d; int l = t % L; int b = t / L;
    int h = l / W, wq = l % W;
    int lt = wq * H + h;
    size_t Md = (size_t)Bb * L * d;
    size_t base = (size_t)b * L * d;
    float t0 = y[0 * Md + base + (size_t)l * d + di];
    float t1 = y[1 * Md + base + (size_t)lt * d + di];
    float t2 = y[2 * Md + base + (size_t)(L - 1 - l) * d + di];
    float t3 = y[3 * Md + base + (size_t)(L - 1 - lt) * d + di];
    ym[base + (size_t)l * d + di] = t0 + t1 + t2 + t3;
}

// ---------------- elementwise kernels ------------------------------------
__global__ void k_gate_silu_h(const float* __restrict__ lnout, const float* __restrict__ xz,
                              _Float16* __restrict__ g, int M, int d, int ld2)
{
    int i = blockIdx.x * blockDim.x + threadIdx.x;
    if (i >= M * d) return;
    int row = i / d, col = i % d;
    float z = xz[(size_t)row * ld2 + d + col];
    float s = z / (1.f + __expf(-z));
    g[i] = (_Float16)(lnout[i] * s);
}

__global__ void k_add(const float* __restrict__ a, const float* __restrict__ b,
                      float* __restrict__ o, int n)
{
    int i = blockIdx.x * blockDim.x + threadIdx.x;
    if (i < n) o[i] = a[i] + b[i];
}

__global__ void k_gelu_add(const float* __restrict__ base, const float* __restrict__ hr,
                           float* __restrict__ dst, int n)
{
    int i = blockIdx.x * blockDim.x + threadIdx.x;
    if (i >= n) return;
    float x = hr[i];
    dst[i] = base[i] + 0.5f * x * (1.f + erff(x * 0.70710678118654752f));
}

__global__ void k_cat2h(const float* __restrict__ a, const float* __restrict__ b,
                        _Float16* __restrict__ o, int M, int C)
{
    int i = blockIdx.x * blockDim.x + threadIdx.x;
    if (i >= M * 2 * C) return;
    int row = i / (2 * C), c = i % (2 * C);
    float v = (c < C) ? a[(size_t)row * C + c] : b[(size_t)row * C + (c - C)];
    o[i] = (_Float16)v;
}

__global__ void k_bn_relu_h(const float* __restrict__ x, const float* __restrict__ g,
                            const float* __restrict__ b, _Float16* __restrict__ o,
                            int M, int C, float eps)
{
    int i = blockIdx.x * blockDim.x + threadIdx.x;
    if (i >= M * C) return;
    int c = i % C;
    float v = x[i] * (g[c] * rsqrtf(1.f + eps)) + b[c];
    o[i] = (_Float16)(v > 0.f ? v : 0.f);
}

__global__ void k_gatecat_h(const float* __restrict__ sarc, const float* __restrict__ optc,
                            const float* __restrict__ graw, _Float16* __restrict__ o,
                            int M, int C)
{
    int i = blockIdx.x * blockDim.x + threadIdx.x;
    if (i >= M * 2 * C) return;
    int row = i / (2 * C), c = i % (2 * C);
    float gate = 1.f / (1.f + __expf(-graw[i]));
    float src = (c < C) ? sarc[(size_t)row * C + c] : optc[(size_t)row * C + (c - C)];
    o[i] = (_Float16)(src * gate);
}

__global__ void k_bn_relu_out(const float* __restrict__ x, const float* __restrict__ g,
                              const float* __restrict__ b, float* __restrict__ o,
                              int M, int C, float eps)
{
    int i = blockIdx.x * blockDim.x + threadIdx.x;
    if (i >= M * C) return;
    int c = i % C;
    float v = x[i] * (g[c] * rsqrtf(1.f + eps)) + b[c];
    o[i] = (v > 0.f ? v : 0.f);
}

// =====================================================================
// Host-side orchestration
// =====================================================================
namespace {

struct Bump { char* base; size_t off; };
static inline void* ba(Bump& b, size_t bytes) {
    size_t a = (b.off + 255) & ~(size_t)255;
    b.off = a + bytes;
    return b.base + a;
}

static inline dim3 ewg(int n) { return dim3((n + 255) / 256); }

static void gemm(const _Float16* A, const _Float16* Bw, float* C, const float* bias,
                 int M, int N, int K, int lda, int ldb, int ldc, hipStream_t s)
{
    int tilesN = N / 16;
    int ntiles = (M / 16) * tilesN;
    const int wpb = 8;                            // 8 waves (tiles) per block
    k_gemm_wmma<<<dim3((ntiles + wpb - 1) / wpb), dim3(32 * wpb), 0, s>>>(
        A, Bw, C, bias, N, K, lda, ldb, ldc, tilesN, ntiles);
}

struct SSW {
    const float *inw, *convw, *convb, *xw, *dtw, *dtb, *alog, *Dp, *lng, *lnb, *outw;
    int ds, ex;
};

static SSW make_ssw(void* const* d_in, int i0, int ds, int ex) {
    SSW w;
    w.inw   = (const float*)d_in[i0 + 0];
    w.convw = (const float*)d_in[i0 + 1];
    w.convb = (const float*)d_in[i0 + 2];
    w.xw    = (const float*)d_in[i0 + 3];
    w.dtw   = (const float*)d_in[i0 + 4];
    w.dtb   = (const float*)d_in[i0 + 5];
    w.alog  = (const float*)d_in[i0 + 6];
    w.Dp    = (const float*)d_in[i0 + 7];
    w.lng   = (const float*)d_in[i0 + 8];
    w.lnb   = (const float*)d_in[i0 + 9];
    w.outw  = (const float*)d_in[i0 + 10];
    w.ds = ds; w.ex = ex;
    return w;
}

// one full SS2D: xin f32 [M,Cm] -> out f32 [M,Cm]; Bump passed by value (arena reuse)
static void run_ss2d(const float* xin, const SSW& w, float* out, Bump ws,
                     hipStream_t s, int Bb, int H, int W, int Cm)
{
    const int L = H * W, M = Bb * L;
    const int d = w.ex * Cm;                  // d_inner: 192 or 96
    const int n = w.ds;                       // 16 or 8
    const int R = 6;                          // ceil(96/16)
    const int Nx = R + 2 * n;                 // 38 or 22
    const int ldx = ((Nx + 15) / 16) * 16;    // 48 or 32 (WMMA N pad)
    const int N2 = 2 * d;
    const int S = 32;                         // scan chunks (L=4096 -> 128 steps)
    const int channels = Bb * 4 * d;

    _Float16* x_h    = (_Float16*)ba(ws, (size_t)M * Cm * 2);
    _Float16* inw_h  = (_Float16*)ba(ws, (size_t)N2 * Cm * 2);
    float*    xz     = (float*)   ba(ws, (size_t)M * N2 * 4);
    float*    xc     = (float*)   ba(ws, (size_t)M * d * 4);
    float*    xs     = (float*)   ba(ws, (size_t)4 * M * d * 4);
    _Float16* xs_h   = (_Float16*)ba(ws, (size_t)4 * M * d * 2);
    _Float16* xw_h   = (_Float16*)ba(ws, (size_t)4 * ldx * d * 2);
    float*    xdbl   = (float*)   ba(ws, (size_t)4 * M * ldx * 4);
    _Float16* xdbl_h = (_Float16*)ba(ws, (size_t)4 * M * ldx * 2);
    _Float16* dtw_h  = (_Float16*)ba(ws, (size_t)4 * d * 32 * 2);
    float*    dts    = (float*)   ba(ws, (size_t)4 * M * d * 4);
    float*    ysc    = (float*)   ba(ws, (size_t)4 * M * d * 4);
    float*    Pb     = (float*)   ba(ws, (size_t)channels * S * n * 4);
    float*    Qb     = (float*)   ba(ws, (size_t)channels * S * n * 4);
    float*    Hin    = (float*)   ba(ws, (size_t)channels * S * n * 4);
    float*    ym     = (float*)   ba(ws, (size_t)M * d * 4);
    float*    lnout  = (float*)   ba(ws, (size_t)M * d * 4);
    _Float16* gat_h  = (_Float16*)ba(ws, (size_t)M * d * 2);
    _Float16* outw_h = (_Float16*)ba(ws, (size_t)Cm * d * 2);

    // f16 conversions (activations unpadded; weights zero-padded to WMMA grain)
    k_w2h<<<ewg(M * Cm), 256, 0, s>>>(xin, x_h, M, Cm, M, Cm);
    k_w2h<<<ewg(N2 * Cm), 256, 0, s>>>(w.inw, inw_h, N2, Cm, N2, Cm);
    for (int k = 0; k < 4; ++k) {
        k_w2h<<<ewg(ldx * d), 256, 0, s>>>(w.xw + (size_t)k * Nx * d,
                                           xw_h + (size_t)k * ldx * d, Nx, d, ldx, d);
        k_w2h<<<ewg(d * 32), 256, 0, s>>>(w.dtw + (size_t)k * d * R,
                                          dtw_h + (size_t)k * d * 32, d, R, d, 32);
    }
    k_w2h<<<ewg(Cm * d), 256, 0, s>>>(w.outw, outw_h, Cm, d, Cm, d);

    // in_proj: xz[M,2d] = x @ in_proj_w^T
    gemm(x_h, inw_h, xz, nullptr, M, N2, Cm, Cm, Cm, N2, s);
    // depthwise conv 3x3 + SiLU on first d columns of xz
    k_dwconv<<<ewg(M * d), 256, 0, s>>>(xz, w.convw, w.convb, xc, Bb, H, W, d, N2);
    // 4 direction copies
    k_build_xs<<<ewg(M * d), 256, 0, s>>>(xc, xs, xs_h, Bb, H, W, d);
    // x_proj per direction: xdbl[k][M,ldx]
    for (int k = 0; k < 4; ++k)
        gemm(xs_h + (size_t)k * M * d, xw_h + (size_t)k * ldx * d,
             xdbl + (size_t)k * M * ldx, nullptr, M, ldx, d, d, d, ldx, s);
    // xdbl -> f16 (dt_proj input; B/C kept f32 for the scan)
    k_w2h<<<ewg(4 * M * ldx), 256, 0, s>>>(xdbl, xdbl_h, 4 * M, ldx, 4 * M, ldx);
    // dt_proj per direction (K padded 6->32, zero weights kill extra columns)
    for (int k = 0; k < 4; ++k)
        gemm(xdbl_h + (size_t)k * M * ldx, dtw_h + (size_t)k * d * 32,
             dts + (size_t)k * M * d, nullptr, M, d, 32, ldx, 32, d, s);

    // chunked selective scan (softplus + dt bias fused; D*u folded in p3)
    if (n == 16) {
        k_scan_p1<16><<<ewg(channels * S), 256, 0, s>>>(xs, xdbl, dts, w.dtb, w.alog,
                                                        Pb, Qb, Bb, L, d, ldx, R, S);
        k_scan_p2<16><<<ewg(channels), 256, 0, s>>>(Pb, Qb, Hin, channels, S);
        k_scan_p3<16><<<ewg(channels * S), 256, 0, s>>>(xs, xdbl, dts, w.dtb, w.alog,
                                                        w.Dp, Hin, ysc,
                                                        Bb, L, d, ldx, R, S);
    } else {
        k_scan_p1<8><<<ewg(channels * S), 256, 0, s>>>(xs, xdbl, dts, w.dtb, w.alog,
                                                       Pb, Qb, Bb, L, d, ldx, R, S);
        k_scan_p2<8><<<ewg(channels), 256, 0, s>>>(Pb, Qb, Hin, channels, S);
        k_scan_p3<8><<<ewg(channels * S), 256, 0, s>>>(xs, xdbl, dts, w.dtb, w.alog,
                                                       w.Dp, Hin, ysc,
                                                       Bb, L, d, ldx, R, S);
    }

    // merge 4 directions
    k_merge<<<ewg(M * d), 256, 0, s>>>(ysc, ym, Bb, H, W, d);
    // LN over d
    k_ln<<<dim3((M + 7) / 8), 256, 0, s>>>(ym, w.lng, w.lnb, lnout, M, d, 1e-5f);
    // * silu(z), to f16
    k_gate_silu_h<<<ewg(M * d), 256, 0, s>>>(lnout, xz, gat_h, M, d, N2);
    // out_proj
    gemm(gat_h, outw_h, out, nullptr, M, Cm, d, d, d, Cm, s);
}

} // namespace

extern "C" void kernel_launch(void* const* d_in, const int* in_sizes, int n_in,
                              void* d_out, int out_size, void* d_ws, size_t ws_size,
                              hipStream_t stream)
{
    (void)in_sizes; (void)n_in; (void)out_size; (void)ws_size;
    const int B = 4, H = 64, W = 64, C = 96;
    const int L = H * W, M = B * L;

    const float* sar_feat = (const float*)d_in[0];
    const float* opt_feat = (const float*)d_in[1];
    // input flattening: dict insertion order, depth-first
    const float* sar_ln1g = (const float*)d_in[2];
    const float* sar_ln1b = (const float*)d_in[3];
    SSW sar_ss = make_ssw(d_in, 4, 16, 2);
    const float* opt_ln1g = (const float*)d_in[15];
    const float* opt_ln1b = (const float*)d_in[16];
    SSW opt_ss = make_ssw(d_in, 17, 16, 2);
    SSW s2o    = make_ssw(d_in, 28, 8, 1);
    SSW o2s    = make_ssw(d_in, 39, 8, 1);
    const float* esar_lng = (const float*)d_in[50];
    const float* esar_lnb = (const float*)d_in[51];
    const float* esar_w   = (const float*)d_in[52];
    const float* esar_b   = (const float*)d_in[53];
    const float* eopt_lng = (const float*)d_in[54];
    const float* eopt_lnb = (const float*)d_in[55];
    const float* eopt_w   = (const float*)d_in[56];
    const float* eopt_b   = (const float*)d_in[57];
    const float* gw1 = (const float*)d_in[58];
    const float* gb1 = (const float*)d_in[59];
    const float* bn1g = (const float*)d_in[60];
    const float* bn1b = (const float*)d_in[61];
    const float* gw2 = (const float*)d_in[62];
    const float* gb2 = (const float*)d_in[63];
    const float* fw  = (const float*)d_in[64];
    const float* fb  = (const float*)d_in[65];
    const float* bn2g = (const float*)d_in[66];
    const float* bn2b = (const float*)d_in[67];

    Bump ws{(char*)d_ws, 0};
    // persistent buffers
    float* sar_e  = (float*)ba(ws, (size_t)M * C * 4);
    float* opt_e  = (float*)ba(ws, (size_t)M * C * 4);
    float* opt_g  = (float*)ba(ws, (size_t)M * C * 4);
    float* sar_g  = (float*)ba(ws, (size_t)M * C * 4);
    float* sar_c  = (float*)ba(ws, (size_t)M * C * 4);
    float* opt_c  = (float*)ba(ws, (size_t)M * C * 4);
    float* tmp_ln = (float*)ba(ws, (size_t)M * C * 4);
    float* tmp_o  = (float*)ba(ws, (size_t)M * C * 4);
    Bump arena = ws;   // scratch reused across the 4 sequential SS2D calls

    // ---- VSS blocks ------------------------------------------------------
    k_ln<<<dim3((M + 7) / 8), 256, 0, stream>>>(sar_feat, sar_ln1g, sar_ln1b,
                                                tmp_ln, M, C, 1e-6f);
    run_ss2d(tmp_ln, sar_ss, tmp_o, arena, stream, B, H, W, C);
    k_add<<<ewg(M * C), 256, 0, stream>>>(sar_feat, tmp_o, sar_e, M * C);

    k_ln<<<dim3((M + 7) / 8), 256, 0, stream>>>(opt_feat, opt_ln1g, opt_ln1b,
                                                tmp_ln, M, C, 1e-6f);
    run_ss2d(tmp_ln, opt_ss, tmp_o, arena, stream, B, H, W, C);
    k_add<<<ewg(M * C), 256, 0, stream>>>(opt_feat, tmp_o, opt_e, M * C);

    // ---- cross SS2D ------------------------------------------------------
    run_ss2d(opt_e, s2o, opt_g, arena, stream, B, H, W, C);
    run_ss2d(sar_e, o2s, sar_g, arena, stream, B, H, W, C);

    // ---- enhance: x_c = x_e + gelu(LN(x_g) @ w^T + b) --------------------
    {
        Bump a2 = arena;
        float*    lno = (float*)   ba(a2, (size_t)M * C * 4);
        _Float16* lnh = (_Float16*)ba(a2, (size_t)M * C * 2);
        _Float16* wh  = (_Float16*)ba(a2, (size_t)C * C * 2);
        float*    hr  = (float*)   ba(a2, (size_t)M * C * 4);
        // sar
        k_ln<<<dim3((M + 7) / 8), 256, 0, stream>>>(sar_g, esar_lng, esar_lnb,
                                                    lno, M, C, 1e-5f);
        k_w2h<<<ewg(M * C), 256, 0, stream>>>(lno, lnh, M, C, M, C);
        k_w2h<<<ewg(C * C), 256, 0, stream>>>(esar_w, wh, C, C, C, C);
        gemm(lnh, wh, hr, esar_b, M, C, C, C, C, C, stream);
        k_gelu_add<<<ewg(M * C), 256, 0, stream>>>(sar_e, hr, sar_c, M * C);
        // opt
        k_ln<<<dim3((M + 7) / 8), 256, 0, stream>>>(opt_g, eopt_lng, eopt_lnb,
                                                    lno, M, C, 1e-5f);
        k_w2h<<<ewg(M * C), 256, 0, stream>>>(lno, lnh, M, C, M, C);
        k_w2h<<<ewg(C * C), 256, 0, stream>>>(eopt_w, wh, C, C, C, C);
        gemm(lnh, wh, hr, eopt_b, M, C, C, C, C, C, stream);
        k_gelu_add<<<ewg(M * C), 256, 0, stream>>>(opt_e, hr, opt_c, M * C);
    }

    // ---- fusion ----------------------------------------------------------
    {
        Bump a2 = arena;
        const int C2 = 2 * C;
        _Float16* cat_h = (_Float16*)ba(a2, (size_t)M * C2 * 2);
        _Float16* gw1h  = (_Float16*)ba(a2, (size_t)C * C2 * 2);
        float*    g1r   = (float*)   ba(a2, (size_t)M * C * 4);
        _Float16* g1h   = (_Float16*)ba(a2, (size_t)M * C * 2);
        _Float16* gw2h  = (_Float16*)ba(a2, (size_t)C2 * C * 2);
        float*    g2r   = (float*)   ba(a2, (size_t)M * C2 * 4);
        _Float16* gch   = (_Float16*)ba(a2, (size_t)M * C2 * 2);
        _Float16* fwh   = (_Float16*)ba(a2, (size_t)C * C2 * 2);
        float*    fr    = (float*)   ba(a2, (size_t)M * C * 4);

        k_cat2h<<<ewg(M * C2), 256, 0, stream>>>(sar_c, opt_c, cat_h, M, C);
        k_w2h<<<ewg(C * C2), 256, 0, stream>>>(gw1, gw1h, C, C2, C, C2);
        gemm(cat_h, gw1h, g1r, gb1, M, C, C2, C2, C2, C, stream);
        k_bn_relu_h<<<ewg(M * C), 256, 0, stream>>>(g1r, bn1g, bn1b, g1h, M, C, 1e-5f);
        k_w2h<<<ewg(C2 * C), 256, 0, stream>>>(gw2, gw2h, C2, C, C2, C);
        gemm(g1h, gw2h, g2r, gb2, M, C2, C, C, C, C2, stream);
        k_gatecat_h<<<ewg(M * C2), 256, 0, stream>>>(sar_c, opt_c, g2r, gch, M, C);
        k_w2h<<<ewg(C * C2), 256, 0, stream>>>(fw, fwh, C, C2, C, C2);
        gemm(gch, fwh, fr, fb, M, C, C2, C2, C2, C, stream);
        k_bn_relu_out<<<ewg(M * C), 256, 0, stream>>>(fr, bn2g, bn2b,
                                                      (float*)d_out, M, C, 1e-5f);
    }
}
